// GPTMoVaE_84241488544009
// MI455X (gfx1250) — matrix operand order repaired and checked by reading, not compiled, not run on hardware
//
#include <hip/hip_runtime.h>

#define N_TOK 8192
#define C_DIM 1024
#define H_DIM 1024
#define E_TOT 12
#define E_MLPC 8
#define V_SZ  32000
#define MT    32          // tokens per MLP tile (2 WMMA M-tiles)
#define LDS_PITCH 1032    // 1024 + 8 bf16 pad (16B-aligned rows, staggered banks)

typedef __attribute__((ext_vector_type(16))) __bf16 v16bf;
typedef __attribute__((ext_vector_type(8)))  float  v8f;

union Frag { v16bf v; unsigned short u[16]; uint4 q[2]; };

__device__ __forceinline__ unsigned short f2bf(float f) {
    unsigned int u = __float_as_uint(f);
    u += 0x7FFFu + ((u >> 16) & 1u);   // round-to-nearest-even
    return (unsigned short)(u >> 16);
}

// ---------------------------------------------------------------------------
// Kernel 0: bulk f32 -> bf16 conversion (one-shot; hoists all conversion VALU
// out of the GEMM hot loop).
// ---------------------------------------------------------------------------
__global__ __launch_bounds__(256) void movae_cvt_bf16_kernel(
    const float* __restrict__ src, unsigned short* __restrict__ dst, int n4)
{
    int i = blockIdx.x * 256 + threadIdx.x;
    if (i >= n4) return;
    float4 v = ((const float4*)src)[i];
    ushort4 h;
    h.x = f2bf(v.x); h.y = f2bf(v.y); h.z = f2bf(v.z); h.w = f2bf(v.w);
    ((ushort4*)dst)[i] = h;
}

// ---------------------------------------------------------------------------
// Kernel 1: routing. One wave32 per token: scores -> softmax -> top2 ->
// compact per-MLP-expert token lists; store VE picks per-token.
// ---------------------------------------------------------------------------
__global__ __launch_bounds__(256) void movae_router_kernel(
    const float* __restrict__ x, const float* __restrict__ router_w,
    float* __restrict__ rweights_out,
    int* __restrict__ counts, int* __restrict__ etok, float* __restrict__ ewt,
    int* __restrict__ tve, float* __restrict__ tvw)
{
    int wave = threadIdx.x >> 5;
    int lane = threadIdx.x & 31;
    int n = blockIdx.x * 8 + wave;
    if (n >= N_TOK) return;

    float s[E_TOT];
    #pragma unroll
    for (int e = 0; e < E_TOT; ++e) s[e] = 0.f;

    const float* xp = x + (size_t)n * C_DIM;
    for (int c = lane; c < C_DIM; c += 32) {
        float xv = xp[c];
        #pragma unroll
        for (int e = 0; e < E_TOT; ++e)
            s[e] = __fmaf_rn(xv, router_w[e * C_DIM + c], s[e]);
    }
    #pragma unroll
    for (int e = 0; e < E_TOT; ++e) {
        #pragma unroll
        for (int off = 16; off >= 1; off >>= 1)
            s[e] += __shfl_xor(s[e], off, 32);
    }
    float m = s[0];
    #pragma unroll
    for (int e = 1; e < E_TOT; ++e) m = fmaxf(m, s[e]);
    float p[E_TOT];
    float denom = 0.f;
    #pragma unroll
    for (int e = 0; e < E_TOT; ++e) { p[e] = __expf(s[e] - m); denom += p[e]; }
    float inv = 1.f / denom;

    if (lane == 0) {
        float* rwp = rweights_out + (size_t)n * E_TOT;
        #pragma unroll
        for (int e = 0; e < E_TOT; ++e) rwp[e] = p[e] * inv;

        int i1 = 0;
        #pragma unroll
        for (int e = 1; e < E_TOT; ++e) if (p[e] > p[i1]) i1 = e;
        int i2 = (i1 == 0) ? 1 : 0;
        #pragma unroll
        for (int e = 0; e < E_TOT; ++e) if (e != i1 && p[e] > p[i2]) i2 = e;

        float w1 = p[i1] * inv, w2 = p[i2] * inv;
        float ns = w1 + w2 + 1e-10f;
        w1 /= ns; w2 /= ns;

        tve[n * 2 + 0] = i1; tvw[n * 2 + 0] = w1;
        tve[n * 2 + 1] = i2; tvw[n * 2 + 1] = w2;
        if (i1 < E_MLPC) {
            int pos = atomicAdd(&counts[i1], 1);
            etok[i1 * N_TOK + pos] = n; ewt[i1 * N_TOK + pos] = w1;
        }
        if (i2 < E_MLPC) {
            int pos = atomicAdd(&counts[i2], 1);
            etok[i2 * N_TOK + pos] = n; ewt[i2 * N_TOK + pos] = w2;
        }
    }
}

// ---------------------------------------------------------------------------
// Kernel 2: vocabulary-embedding contribution; zero-initializes every output
// row so the MLP kernel can accumulate with atomics.
// ---------------------------------------------------------------------------
__global__ __launch_bounds__(256) void movae_ve_kernel(
    const int* __restrict__ token_ids, const float* __restrict__ ve_tables,
    const int* __restrict__ tve, const float* __restrict__ tvw,
    float* __restrict__ out)
{
    int n = blockIdx.x;
    int tid = token_ids[n];
    float4 acc = make_float4(0.f, 0.f, 0.f, 0.f);
    #pragma unroll
    for (int k = 0; k < 2; ++k) {
        int e = tve[n * 2 + k];
        if (e >= E_MLPC) {
            float w = tvw[n * 2 + k];
            const float4* row = (const float4*)(ve_tables +
                ((size_t)(e - E_MLPC) * V_SZ + tid) * C_DIM);
            float4 v = row[threadIdx.x];
            acc.x = __fmaf_rn(w, v.x, acc.x);
            acc.y = __fmaf_rn(w, v.y, acc.y);
            acc.z = __fmaf_rn(w, v.z, acc.z);
            acc.w = __fmaf_rn(w, v.w, acc.w);
        }
    }
    ((float4*)(out + (size_t)n * C_DIM))[threadIdx.x] = acc;
}

// ---------------------------------------------------------------------------
// Kernel 3: routed expert MLP via v_wmma_f32_16x16x32_bf16, bf16 operands
// preconverted. Block = (expert, 32-token tile): 2 A-tiles share each B
// fragment; 8 waves cover the full H / C dimension. x tile staged to LDS via
// gfx1250 async-to-LDS path.
// ---------------------------------------------------------------------------
__global__ __launch_bounds__(256) void movae_mlp_kernel(
    const unsigned short* __restrict__ xbf,
    const unsigned short* __restrict__ fcbf,
    const unsigned short* __restrict__ pjbf,
    float* __restrict__ out,
    const int* __restrict__ counts, const int* __restrict__ etok,
    const float* __restrict__ ewt)
{
    int e = blockIdx.y;
    int cnt = counts[e];
    int tbase = blockIdx.x * MT;
    if (tbase >= cnt) return;

    __shared__ unsigned short ldsA[MT][LDS_PITCH];  // x tile, then hidden tile
    __shared__ int   toks[MT];
    __shared__ float wts[MT];

    int tidx = threadIdx.x;
    int wave = tidx >> 5;
    int lane = tidx & 31;
    int half = lane >> 4;
    int l15  = lane & 15;

    if (tidx < MT) {
        int idx = tbase + tidx;
        bool valid = idx < cnt;
        int src = valid ? idx : tbase;        // pad rows duplicate first token
        toks[tidx] = etok[e * N_TOK + src];
        wts[tidx]  = valid ? ewt[e * N_TOK + idx] : 0.f;
    }
    __syncthreads();

    // --- async copy x tile (32 x 1024 bf16 = 64KB) -> LDS ---
    // Each lane moves 16 x 16B; immediate offset applies to BOTH the LDS and
    // the global address (rows are contiguous on both sides).
    {
        int row = tidx >> 3;          // 8 threads per row
        int off = (tidx & 7) * 128;   // ushort elements
        unsigned lds_addr = (unsigned)(unsigned long long)&ldsA[row][off];
        unsigned long long gaddr =
            (unsigned long long)(xbf + (size_t)toks[row] * C_DIM + off);
        asm volatile(
            "global_load_async_to_lds_b128 %0, %1, off\n\t"
            "global_load_async_to_lds_b128 %0, %1, off offset:16\n\t"
            "global_load_async_to_lds_b128 %0, %1, off offset:32\n\t"
            "global_load_async_to_lds_b128 %0, %1, off offset:48\n\t"
            "global_load_async_to_lds_b128 %0, %1, off offset:64\n\t"
            "global_load_async_to_lds_b128 %0, %1, off offset:80\n\t"
            "global_load_async_to_lds_b128 %0, %1, off offset:96\n\t"
            "global_load_async_to_lds_b128 %0, %1, off offset:112\n\t"
            "global_load_async_to_lds_b128 %0, %1, off offset:128\n\t"
            "global_load_async_to_lds_b128 %0, %1, off offset:144\n\t"
            "global_load_async_to_lds_b128 %0, %1, off offset:160\n\t"
            "global_load_async_to_lds_b128 %0, %1, off offset:176\n\t"
            "global_load_async_to_lds_b128 %0, %1, off offset:192\n\t"
            "global_load_async_to_lds_b128 %0, %1, off offset:208\n\t"
            "global_load_async_to_lds_b128 %0, %1, off offset:224\n\t"
            "global_load_async_to_lds_b128 %0, %1, off offset:240\n\t"
            :
            : "v"(lds_addr), "v"(gaddr)
            : "memory");
        asm volatile("s_wait_asynccnt 0" ::: "memory");
    }
    __syncthreads();

    v8f acc0[8], acc1[8];
    #pragma unroll
    for (int j = 0; j < 8; ++j)
        #pragma unroll
        for (int r = 0; r < 8; ++r) { acc0[j][r] = 0.f; acc1[j][r] = 0.f; }

    // lane-invariant parts of B addressing folded into the base pointer:
    // element = e*H*C + (tile_row16 + l15)*C + kbase + 8*half
    const unsigned short* Bfc = fcbf + (size_t)e * H_DIM * C_DIM
                              + (size_t)(wave * 8 * 16 + l15) * C_DIM + 8 * half;

    // --- stage A: hidden = relu(x @ fc^T)^2, K = C ---
    for (int kc = 0; kc < C_DIM / 32; ++kc) {
        int kbase = kc * 32;
        int kb0 = kbase + 8 * half;
        Frag a0, a1;
        a0.q[0] = *(const uint4*)&ldsA[l15][kb0];
        a0.q[1] = *(const uint4*)&ldsA[l15][kb0 + 16];
        a1.q[0] = *(const uint4*)&ldsA[16 + l15][kb0];
        a1.q[1] = *(const uint4*)&ldsA[16 + l15][kb0 + 16];
        __builtin_prefetch(Bfc + kbase + 256, 0, 3);   // WGP-scope prefetch
        #pragma unroll
        for (int j = 0; j < 8; ++j) {
            const uint4* bq = (const uint4*)(Bfc + (size_t)j * 16 * C_DIM + kbase);
            Frag b;
            b.q[0] = bq[0];
            b.q[1] = bq[2];
            acc0[j] = __builtin_amdgcn_wmma_f32_16x16x32_bf16(
                false, a0.v, false, b.v, (short)0, acc0[j], false, false);
            acc1[j] = __builtin_amdgcn_wmma_f32_16x16x32_bf16(
                false, a1.v, false, b.v, (short)0, acc1[j], false, false);
        }
    }
    __syncthreads();   // all x reads done; reuse LDS for hidden

    // --- hidden (relu^2) -> bf16 LDS ---
    #pragma unroll
    for (int j = 0; j < 8; ++j) {
        int hcol = (wave * 8 + j) * 16 + l15;
        #pragma unroll
        for (int r = 0; r < 8; ++r) {
            int M = r + 8 * half;               // C/D layout: M = r + 8*(lane>=16)
            float v0 = acc0[j][r];
            float v1 = acc1[j][r];
            v0 = v0 > 0.f ? v0 * v0 : 0.f;
            v1 = v1 > 0.f ? v1 * v1 : 0.f;
            ldsA[M][hcol]      = f2bf(v0);
            ldsA[16 + M][hcol] = f2bf(v1);
        }
    }
    __syncthreads();

    // --- stage B: out = hidden @ proj^T, K = H ---
    #pragma unroll
    for (int j = 0; j < 8; ++j)
        #pragma unroll
        for (int r = 0; r < 8; ++r) { acc0[j][r] = 0.f; acc1[j][r] = 0.f; }

    const unsigned short* Bpj = pjbf + (size_t)e * C_DIM * H_DIM
                              + (size_t)(wave * 8 * 16 + l15) * H_DIM + 8 * half;

    for (int kc = 0; kc < H_DIM / 32; ++kc) {
        int kbase = kc * 32;
        int kb0 = kbase + 8 * half;
        Frag a0, a1;
        a0.q[0] = *(const uint4*)&ldsA[l15][kb0];
        a0.q[1] = *(const uint4*)&ldsA[l15][kb0 + 16];
        a1.q[0] = *(const uint4*)&ldsA[16 + l15][kb0];
        a1.q[1] = *(const uint4*)&ldsA[16 + l15][kb0 + 16];
        __builtin_prefetch(Bpj + kbase + 256, 0, 3);
        #pragma unroll
        for (int j = 0; j < 8; ++j) {
            const uint4* bq = (const uint4*)(Bpj + (size_t)j * 16 * H_DIM + kbase);
            Frag b;
            b.q[0] = bq[0];
            b.q[1] = bq[2];
            acc0[j] = __builtin_amdgcn_wmma_f32_16x16x32_bf16(
                false, a0.v, false, b.v, (short)0, acc0[j], false, false);
            acc1[j] = __builtin_amdgcn_wmma_f32_16x16x32_bf16(
                false, a1.v, false, b.v, (short)0, acc1[j], false, false);
        }
    }

    // --- weighted scatter into output ---
    #pragma unroll
    for (int j = 0; j < 8; ++j) {
        int cc = (wave * 8 + j) * 16 + l15;
        #pragma unroll
        for (int r = 0; r < 8; ++r) {
            int M = r + 8 * half;
            float w0 = wts[M];
            float w1 = wts[16 + M];
            if (w0 != 0.f)
                atomicAdd(&out[(size_t)toks[M] * C_DIM + cc], w0 * acc0[j][r]);
            if (w1 != 0.f)
                atomicAdd(&out[(size_t)toks[16 + M] * C_DIM + cc], w1 * acc1[j][r]);
        }
    }
}

// ---------------------------------------------------------------------------
extern "C" void kernel_launch(void* const* d_in, const int* in_sizes, int n_in,
                              void* d_out, int out_size, void* d_ws, size_t ws_size,
                              hipStream_t stream) {
    const float* x         = (const float*)d_in[0];
    const int*   token_ids = (const int*)  d_in[1];
    const float* router_w  = (const float*)d_in[2];
    const float* fc_w      = (const float*)d_in[3];
    const float* proj_w    = (const float*)d_in[4];
    const float* ve_tables = (const float*)d_in[5];

    float* out      = (float*)d_out;                      // [N, C]
    float* rweights = out + (size_t)N_TOK * C_DIM;        // [N, 12]

    char* ws = (char*)d_ws;
    int*   counts = (int*)  ws;                                           // 8 ints
    int*   tve    = (int*)  (ws + 256);                                   // N*2 int
    float* tvw    = (float*)(ws + 256 + N_TOK * 2 * 4);                   // N*2 f32
    int*   etok   = (int*)  (ws + 256 + N_TOK * 4 * 4);                   // 8*N int
    float* ewt    = (float*)(ws + 256 + N_TOK * 4 * 4 + E_MLPC * N_TOK * 4);

    const size_t XN  = (size_t)N_TOK * C_DIM;             // 8.39M elements
    const size_t WN  = (size_t)E_MLPC * H_DIM * C_DIM;    // 8.39M elements
    unsigned short* xbf  = (unsigned short*)(ws + (1u << 20));
    unsigned short* fcbf = xbf + XN;
    unsigned short* pjbf = fcbf + WN;

    hipMemsetAsync(counts, 0, E_MLPC * sizeof(int), stream);

    movae_cvt_bf16_kernel<<<(int)(XN / 4 / 256), 256, 0, stream>>>(x, xbf, (int)(XN / 4));
    movae_cvt_bf16_kernel<<<(int)(WN / 4 / 256), 256, 0, stream>>>(fc_w, fcbf, (int)(WN / 4));
    movae_cvt_bf16_kernel<<<(int)(WN / 4 / 256), 256, 0, stream>>>(proj_w, pjbf, (int)(WN / 4));

    movae_router_kernel<<<N_TOK / 8, 256, 0, stream>>>(
        x, router_w, rweights, counts, etok, ewt, tve, tvw);

    movae_ve_kernel<<<N_TOK, 256, 0, stream>>>(
        token_ids, ve_tables, tve, tvw, out);

    movae_mlp_kernel<<<dim3(N_TOK / MT, E_MLPC), 256, 0, stream>>>(
        xbf, fcbf, pjbf, out, counts, etok, ewt);
}
